// MultiHeadTransformer_18554258719372
// MI455X (gfx1250) — matrix operand order repaired and checked by reading, not compiled, not run on hardware
//
#include <hip/hip_runtime.h>
#include <hip/hip_bf16.h>

// ---------------- problem constants ----------------
#define NB      8
#define NCIN    256
#define NCI     128
#define NHW     64
#define NVIEWS  4
#define NT      4
#define NV      17
#define NP      272                     // NVIEWS*NT*NV
#define NCOL    17408                   // NHW*NP  (columns per batch in GEMM layout)
#define TOTE    ((size_t)NB*NCIN*NCOL)  // 35,651,584 elements
#define GN_INV  (1.0f/4456448.0f)       // 1/(NCIN*NCOL)
#define FF_INV  (1.0f/16384.0f)         // 1/(NCIN*NHW)

typedef __attribute__((ext_vector_type(16))) __bf16 v16bf;
typedef __attribute__((ext_vector_type(8)))  float  v8f;

__constant__ int c_partJoints[17] = {11,12,13, 14,15,16, 4,5,6, 1,2,3, 0,7,8,9,10};
__constant__ int c_partStart[5]   = {0,3,6,9,12};
__constant__ int c_partSize[5]    = {3,3,3,3,5};
__constant__ int c_partOfJoint[17]= {4,3,3,3,2,2,2,4,4,4,4,0,0,0,1,1,1};
__constant__ int c_jpOfJoint[17]  = {0,0,1,2,0,1,2,1,2,3,4,0,1,2,0,1,2};
__constant__ int c_aoff[5]        = {0,4608,9216,13824,18432};   // attnP float offsets
__constant__ int c_vbase[5]       = {0,3072,6144,9216,12288};    // per-part column bases

// ---------------- weight prep ----------------
// CC[h][o][c] = sum_ci W_w[o][slice(h)+ci] * g_w[h][ci][c]   (bf16, 7 matrices 256x256)
__global__ void k_combined_w(const float* __restrict__ W_w, const float* __restrict__ g_w,
                             __hip_bfloat16* __restrict__ CC) {
  int h = blockIdx.y, o = blockIdx.x, c = threadIdx.x;
  int slice = (h == 0) ? 0 : (h == 1) ? 128 : 256;
  float s = 0.f;
  for (int ci = 0; ci < 128; ++ci)
    s += W_w[o*384 + slice + ci] * g_w[((size_t)(h*128 + ci))*256 + c];
  CC[((size_t)(h*256 + o))*256 + c] = __float2bfloat16(s);
}

__global__ void k_cvt_bf16(const float* __restrict__ src, __hip_bfloat16* __restrict__ dst, int n) {
  int i = blockIdx.x*256 + threadIdx.x;
  if (i < n) dst[i] = __float2bfloat16(src[i]);
}

// zbias[o][j] = W_b[o] + Wj@g_b0 + Wt@g_b1 + Wv@g_b[2+part(j)]
__global__ void k_zbias(const float* __restrict__ W_w, const float* __restrict__ g_b,
                        const float* __restrict__ W_b, float* __restrict__ zbias) {
  int o = threadIdx.x;
  float s01 = W_b[o];
  for (int ci = 0; ci < 128; ++ci)
    s01 += W_w[o*384 + ci]*g_b[ci] + W_w[o*384 + 128 + ci]*g_b[128 + ci];
  float sv[5];
  for (int p = 0; p < 5; ++p) {
    float s = 0.f;
    for (int ci = 0; ci < 128; ++ci)
      s += W_w[o*384 + 256 + ci] * g_b[(2 + p)*128 + ci];
    sv[p] = s;
  }
  for (int j = 0; j < 17; ++j) zbias[o*17 + j] = s01 + sv[c_partOfJoint[j]];
}

// ---------------- pooling + theta/phi projections ----------------
__global__ void k_pool(const float* __restrict__ x, float* __restrict__ pool) {
  int bc = blockIdx.x;                        // b*256 + c
  const float* xp = x + (size_t)bc * NCOL;
  for (int p = threadIdx.x; p < NP; p += blockDim.x) {
    float s = 0.f;
    for (int hw = 0; hw < NHW; ++hw) s += xp[hw*NP + p];
    pool[(size_t)bc*NP + p] = s * (1.f/64.f);
  }
}

// TH/PH[h][b][o][p] for all 7 heads over all 272 pooled columns
__global__ void k_thph(const float* __restrict__ pool,
                       const float* __restrict__ tw, const float* __restrict__ tb,
                       const float* __restrict__ pw, const float* __restrict__ pb,
                       float* __restrict__ TH, float* __restrict__ PH) {
  int p = blockIdx.x, hb = blockIdx.y;
  int h = hb >> 3, b = hb & 7;
  __shared__ float col[256];
  int o = threadIdx.x;                        // 128 threads
  col[o]       = pool[((size_t)(b*256) + o)      *NP + p];
  col[o + 128] = pool[((size_t)(b*256) + o + 128)*NP + p];
  __syncthreads();
  float st = tb[h*128 + o], sp = pb[h*128 + o];
  const float* twp = tw + (size_t)(h*128 + o)*256;
  const float* pwp = pw + (size_t)(h*128 + o)*256;
  #pragma unroll 8
  for (int c = 0; c < 256; ++c) { st += twp[c]*col[c]; sp += pwp[c]*col[c]; }
  size_t idx = ((size_t)(h*8 + b)*128 + o)*NP + p;
  TH[idx] = st; PH[idx] = sp;
}

// ---------------- softmaxes ----------------
__global__ void k_attnJ(const float* __restrict__ TH, const float* __restrict__ PH,
                        float* __restrict__ attnJ) {
  int n = blockIdx.x;                         // (b,v,t): b*16 + v*4 + t
  int b = n >> 4, v = (n >> 2) & 3, t = n & 3;
  __shared__ float thc[17][128], phc[17][128], lg[17][17];
  int tid = threadIdx.x;                      // 128
  size_t base = ((size_t)b*128)*NP;           // h=0
  for (int s = 0; s < 17; ++s) {
    int p = (v*NT + t)*NV + s;
    thc[s][tid] = TH[base + (size_t)tid*NP + p];
    phc[s][tid] = PH[base + (size_t)tid*NP + p];
  }
  __syncthreads();
  for (int e = tid; e < 289; e += 128) {
    int q = e / 17, k = e % 17;
    float s = 0.f;
    for (int o = 0; o < 128; ++o) s += thc[q][o]*phc[k][o];
    lg[q][k] = s;
  }
  __syncthreads();
  if (tid < 17) {
    float mx = -1e30f;
    for (int k = 0; k < 17; ++k) mx = fmaxf(mx, lg[tid][k]);
    float e_[17], sm = 0.f;
    for (int k = 0; k < 17; ++k) { e_[k] = __expf(lg[tid][k] - mx); sm += e_[k]; }
    float inv = 1.f / sm;
    for (int k = 0; k < 17; ++k) attnJ[((size_t)n*17 + tid)*17 + k] = e_[k]*inv;
  }
}

__global__ void k_attnT(const float* __restrict__ TH, const float* __restrict__ PH,
                        float* __restrict__ attnT) {
  int n = blockIdx.x;                         // (b,v,j): (b*4+v)*17 + j
  int b = n / 68, r = n % 68, v = r / 17, j = r % 17;
  __shared__ float thc[4][128], phc[4][128], lg[16];
  int tid = threadIdx.x;                      // 128
  size_t base = ((size_t)(8 + b)*128)*NP;     // h=1
  for (int s = 0; s < 4; ++s) {
    int p = (v*NT + s)*NV + j;
    thc[s][tid] = TH[base + (size_t)tid*NP + p];
    phc[s][tid] = PH[base + (size_t)tid*NP + p];
  }
  __syncthreads();
  if (tid < 16) {
    int q = tid >> 2, k = tid & 3;
    float s = 0.f;
    for (int o = 0; o < 128; ++o) s += thc[q][o]*phc[k][o];
    lg[tid] = s;
  }
  __syncthreads();
  if (tid < 4) {
    float mx = -1e30f;
    for (int k = 0; k < 4; ++k) mx = fmaxf(mx, lg[tid*4 + k]);
    float e_[4], sm = 0.f;
    for (int k = 0; k < 4; ++k) { e_[k] = __expf(lg[tid*4 + k] - mx); sm += e_[k]; }
    float inv = 1.f / sm;
    for (int k = 0; k < 4; ++k) attnT[((size_t)n*4 + tid)*4 + k] = e_[k]*inv;
  }
}

__global__ void k_attnP(const float* __restrict__ TH, const float* __restrict__ PH,
                        float* __restrict__ attnP) {
  int bt = blockIdx.x;                        // b*4 + t
  int b = bt >> 2, t = bt & 3, p = blockIdx.y;
  int Vp = c_partSize[p], S = 4*Vp;
  __shared__ float thc[20][128], phc[20][128], lg[20][20];
  int tid = threadIdx.x;                      // 256
  size_t base = ((size_t)((2 + p)*8 + b)*128)*NP;
  for (int s = tid >> 7; s < S; s += 2) {
    int o = tid & 127;
    int v = s / Vp, jp = s % Vp;
    int j = c_partJoints[c_partStart[p] + jp];
    int pc = (v*NT + t)*NV + j;
    thc[s][o] = TH[base + (size_t)o*NP + pc];
    phc[s][o] = PH[base + (size_t)o*NP + pc];
  }
  __syncthreads();
  for (int e = tid; e < S*S; e += 256) {
    int q = e / S, k = e % S;
    float s = 0.f;
    for (int o = 0; o < 128; ++o) s += thc[q][o]*phc[k][o];
    lg[q][k] = s;
  }
  __syncthreads();
  if (tid < S) {
    float mx = -1e30f;
    for (int k = 0; k < S; ++k) mx = fmaxf(mx, lg[tid][k]);
    float sm = 0.f;
    for (int k = 0; k < S; ++k) sm += __expf(lg[tid][k] - mx);
    float inv = 1.f / sm;
    size_t ob = (size_t)c_aoff[p] + ((size_t)bt*S + tid)*S;
    for (int k = 0; k < S; ++k) attnP[ob + k] = __expf(lg[tid][k] - mx)*inv;
  }
}

// ---------------- attention mixing (writes bf16 GEMM inputs) ----------------
__global__ void k_mixJ(const float* __restrict__ x, const float* __restrict__ attnJ,
                       __hip_bfloat16* __restrict__ Xj) {
  int g = blockIdx.x; int hw = g & 63; int bc = g >> 6; int b = bc >> 8;
  __shared__ float row[NP];
  const float* xp = x + (size_t)bc*NCOL + hw*NP;
  for (int p = threadIdx.x; p < NP; p += 128) row[p] = xp[p];
  __syncthreads();
  __hip_bfloat16* op = Xj + (size_t)bc*NCOL + hw*NP;
  for (int p = threadIdx.x; p < NP; p += 128) {
    int q = p % NV, vt = p / NV;
    const float* a = attnJ + ((size_t)(b*16 + vt)*17 + q)*17;
    float s = 0.f;
    #pragma unroll
    for (int k = 0; k < 17; ++k) s += a[k]*row[vt*NV + k];
    op[p] = __float2bfloat16(s);
  }
}

__global__ void k_mixT(const float* __restrict__ x, const float* __restrict__ attnT,
                       __hip_bfloat16* __restrict__ Xt) {
  int g = blockIdx.x; int hw = g & 63; int bc = g >> 6; int b = bc >> 8;
  __shared__ float row[NP];
  const float* xp = x + (size_t)bc*NCOL + hw*NP;
  for (int p = threadIdx.x; p < NP; p += 128) row[p] = xp[p];
  __syncthreads();
  __hip_bfloat16* op = Xt + (size_t)bc*NCOL + hw*NP;
  for (int p = threadIdx.x; p < NP; p += 128) {
    int j = p % NV, vt = p / NV, v = vt >> 2, q = vt & 3;
    const float* a = attnT + ((size_t)((b*4 + v)*17 + j)*4 + q)*4;
    float s = 0.f;
    #pragma unroll
    for (int k = 0; k < 4; ++k) s += a[k]*row[(v*4 + k)*NV + j];
    op[p] = __float2bfloat16(s);
  }
}

__global__ void k_mixV(const float* __restrict__ x, const float* __restrict__ attnP,
                       __hip_bfloat16* __restrict__ Xv) {
  int g = blockIdx.x; int hw = g & 63; int bc = g >> 6; int b = bc >> 8;
  __shared__ float row[NP];
  const float* xp = x + (size_t)bc*NCOL + hw*NP;
  for (int p = threadIdx.x; p < NP; p += 128) row[p] = xp[p];
  __syncthreads();
  for (int u = threadIdx.x; u < NP; u += 128) {
    int pt, r;
    if (u < 192) { pt = u / 48; r = u % 48; } else { pt = 4; r = u - 192; }
    int Vp = c_partSize[pt], S = 4*Vp;
    int t = r / (4*Vp); int rr = r % (4*Vp); int v = rr / Vp; int jp = rr % Vp;
    int s = v*Vp + jp;
    const float* a = attnP + c_aoff[pt] + ((size_t)(b*4 + t)*S + s)*S;
    float acc = 0.f;
    for (int k = 0; k < S; ++k) {
      int vk = k / Vp, jpk = k % Vp;
      int jg = c_partJoints[c_partStart[pt] + jpk];
      acc += a[k]*row[(vk*4 + t)*NV + jg];
    }
    int col = c_vbase[pt] + ((hw*4 + t)*4 + v)*Vp + jp;
    Xv[(size_t)bc*NCOL + col] = __float2bfloat16(acc);
  }
}

// ---------------- WMMA bf16 GEMM: C[b] (+)= A(256x256) x Bmat[b](256 x ncols) ----------------
__global__ __launch_bounds__(256) void k_gemm256(
    const __hip_bfloat16* __restrict__ Am, const __hip_bfloat16* __restrict__ Bm,
    float* __restrict__ Cm, const float* __restrict__ bias,
    int ldb, int ldc, long long strideB, long long strideC, int accFlag)
{
  const int bz = blockIdx.z;
  const int col0 = blockIdx.x * 64;
  const int tid = threadIdx.x;
  const int w = tid >> 5, lane = tid & 31;
  const int nt = w & 3, mh = w >> 2;
  const int mbase = blockIdx.y * 128 + mh * 64;   // this wave: mtiles mbase/16 .. +3
  const int n = col0 + nt * 16;
  const int lm = lane & 15, kh = lane >> 4;

  const __hip_bfloat16* Bp = Bm + (size_t)bz * strideB;
  float* Cp = Cm + (size_t)bz * strideC;

  v8f acc[4];
  #pragma unroll
  for (int mi = 0; mi < 4; ++mi) {
    if (accFlag) {
      #pragma unroll
      for (int r = 0; r < 8; ++r)
        acc[mi][r] = Cp[(size_t)(mbase + mi*16 + r + 8*kh)*ldc + n + lm];
    } else {
      #pragma unroll
      for (int r = 0; r < 8; ++r) acc[mi][r] = 0.0f;
    }
  }

  #pragma unroll
  for (int kb = 0; kb < 256; kb += 32) {
    // B fragment: lane holds column (n+lm), 16 k values in table order
    union { unsigned short u[16]; v16bf v; } bf;
    #pragma unroll
    for (int s = 0; s < 16; ++s) {
      const int k = kb + ((s >> 3) << 4) + (kh << 3) + (s & 7);
      bf.u[s] = *(const unsigned short*)(Bp + (size_t)k*ldb + n + lm);
    }
    if (kb + 32 < 256)
      __builtin_prefetch((const void*)(Bp + (size_t)(kb + 32 + lane)*ldb + n), 0, 3);
    #pragma unroll
    for (int mi = 0; mi < 4; ++mi) {
      const int m = mbase + mi*16 + lm;
      union { unsigned u[8]; v16bf v; } af;
      #pragma unroll
      for (int i = 0; i < 8; ++i) {
        const int k = kb + ((i >> 2) << 4) + (kh << 3) + ((i & 3) << 1);
        af.u[i] = *(const unsigned*)(Am + m*256 + k);
      }
      acc[mi] = __builtin_amdgcn_wmma_f32_16x16x32_bf16(
          false, af.v, false, bf.v, (short)0, acc[mi], false, false);
    }
  }

  #pragma unroll
  for (int mi = 0; mi < 4; ++mi) {
    #pragma unroll
    for (int r = 0; r < 8; ++r) {
      const int m = mbase + mi*16 + r + 8*kh;
      float v = acc[mi][r];
      if (bias) v += bias[m];
      Cp[(size_t)m*ldc + n + lm] = v;
    }
  }
}

// ---------------- combine heads + bias + residual (in place into zacc) ----------------
__global__ void k_combine(float* __restrict__ z, const float* __restrict__ zV,
                          const float* __restrict__ zbias, const float* __restrict__ x) {
  int g = blockIdx.x; int hw = g & 63; int bc = g >> 6; int o = bc & 255;
  size_t base = (size_t)bc*NCOL + hw*NP;
  for (int p = threadIdx.x; p < NP; p += 128) {
    int j = p % NV, vt = p / NV, v = vt >> 2, t = vt & 3;
    int pp = c_partOfJoint[j], jp = c_jpOfJoint[j];
    int Vp = c_partSize[pp];
    int col = c_vbase[pp] + ((hw*4 + t)*4 + v)*Vp + jp;
    z[base + p] = z[base + p] + zV[(size_t)bc*NCOL + col] + zbias[o*17 + j] + x[base + p];
  }
}

// ---------------- GroupNorm machinery ----------------
__global__ void k_zero(float* __restrict__ p, int n) {
  int i = blockIdx.x*256 + threadIdx.x;
  if (i < n) p[i] = 0.f;
}

__global__ void k_gstats(const float* __restrict__ z, float* __restrict__ gstat) {
  int bc = blockIdx.x; int b = bc >> 8;
  const float* zp = z + (size_t)bc*NCOL;
  float s = 0.f, q = 0.f;
  for (int i = threadIdx.x; i < NCOL; i += 256) { float v = zp[i]; s += v; q += v*v; }
  __shared__ float rs[256], rq[256];
  rs[threadIdx.x] = s; rq[threadIdx.x] = q;
  __syncthreads();
  for (int st = 128; st > 0; st >>= 1) {
    if (threadIdx.x < st) { rs[threadIdx.x] += rs[threadIdx.x + st]; rq[threadIdx.x] += rq[threadIdx.x + st]; }
    __syncthreads();
  }
  if (threadIdx.x == 0) { atomicAdd(&gstat[b], rs[0]); atomicAdd(&gstat[8 + b], rq[0]); }
}

// normalize z in place (f32) and emit bf16 copy for the ff1 GEMM
__global__ void k_gnorm_dual(float* __restrict__ z, const float* __restrict__ gstat,
                             const float* __restrict__ w, const float* __restrict__ bb,
                             __hip_bfloat16* __restrict__ znb) {
  size_t i = (size_t)blockIdx.x*256 + threadIdx.x;
  size_t bc = i / NCOL; int b = (int)(bc >> 8), c = (int)(bc & 255);
  int hw = (int)((i % NCOL) / NP);
  float mu = gstat[b]*GN_INV;
  float var = gstat[8 + b]*GN_INV - mu*mu;
  float rv = rsqrtf(var + 1e-5f);
  float vn = (z[i] - mu)*rv*w[c*NHW + hw] + bb[c*NHW + hw];
  z[i] = vn;
  znb[i] = __float2bfloat16(vn);
}

__global__ void k_gnorm_out(const float* __restrict__ h, const float* __restrict__ gstat,
                            const float* __restrict__ w, const float* __restrict__ bb,
                            float* __restrict__ out) {
  size_t i = (size_t)blockIdx.x*256 + threadIdx.x;
  size_t bc = i / NCOL; int b = (int)(bc >> 8), c = (int)(bc & 255);
  int hw = (int)((i % NCOL) / NP);
  float mu = gstat[b]*GN_INV;
  float var = gstat[8 + b]*GN_INV - mu*mu;
  float rv = rsqrtf(var + 1e-5f);
  out[i] = (h[i] - mu)*rv*w[c*NHW + hw] + bb[c*NHW + hw];
}

// ---------------- FFN layernorm over (c,hw) per position ----------------
__global__ void k_ffstats1(const float* __restrict__ h1, float* __restrict__ pS,
                           float* __restrict__ pQ) {
  int bc = blockIdx.x;
  const float* hp = h1 + (size_t)bc*NCOL;
  for (int p = threadIdx.x; p < NP; p += 256) {
    float s = 0.f, q = 0.f;
    for (int hw = 0; hw < NHW; ++hw) { float v = hp[hw*NP + p]; s += v; q += v*v; }
    pS[(size_t)bc*NP + p] = s; pQ[(size_t)bc*NP + p] = q;
  }
}

__global__ void k_ffstats2(const float* __restrict__ pS, const float* __restrict__ pQ,
                           float* __restrict__ mu, float* __restrict__ rv) {
  int bp = blockIdx.x; int b = bp / NP; int p = bp % NP;
  __shared__ float rs[256], rq[256];
  int c = threadIdx.x;
  rs[c] = pS[((size_t)(b*256 + c))*NP + p];
  rq[c] = pQ[((size_t)(b*256 + c))*NP + p];
  __syncthreads();
  for (int st = 128; st > 0; st >>= 1) {
    if (c < st) { rs[c] += rs[c + st]; rq[c] += rq[c + st]; }
    __syncthreads();
  }
  if (c == 0) {
    float m = rs[0]*FF_INV;
    float v = rq[0]*FF_INV - m*m;
    mu[bp] = m; rv[bp] = rsqrtf(v + 1e-5f);
  }
}

__global__ void k_ffnorm(const float* __restrict__ h1, const float* __restrict__ mu,
                         const float* __restrict__ rv, const float* __restrict__ fw,
                         const float* __restrict__ fb, __hip_bfloat16* __restrict__ outb) {
  size_t i = (size_t)blockIdx.x*256 + threadIdx.x;
  size_t bc = i / NCOL; int b = (int)(bc >> 8), c = (int)(bc & 255);
  int rem = (int)(i % NCOL); int hw = rem / NP; int p = rem % NP;
  float v = (h1[i] - mu[b*NP + p])*rv[b*NP + p]*fw[c*NHW + hw] + fb[c*NHW + hw];
  outb[i] = __float2bfloat16(fmaxf(v, 0.f));
}

__global__ void k_add(float* __restrict__ a, const float* __restrict__ b2) {
  size_t i = (size_t)blockIdx.x*256 + threadIdx.x;
  a[i] += b2[i];
}

// ---------------- host driver ----------------
extern "C" void kernel_launch(void* const* d_in, const int* in_sizes, int n_in,
                              void* d_out, int out_size, void* d_ws, size_t ws_size,
                              hipStream_t stream) {
  (void)in_sizes; (void)n_in; (void)out_size; (void)ws_size;
  const float* x      = (const float*)d_in[0];
  const float* g_w    = (const float*)d_in[1];
  const float* g_b    = (const float*)d_in[2];
  const float* th_w   = (const float*)d_in[3];
  const float* th_b   = (const float*)d_in[4];
  const float* ph_w   = (const float*)d_in[5];
  const float* ph_b   = (const float*)d_in[6];
  const float* W_w    = (const float*)d_in[7];
  const float* W_b    = (const float*)d_in[8];
  const float* ln1_w  = (const float*)d_in[9];
  const float* ln1_b  = (const float*)d_in[10];
  const float* ff1_w  = (const float*)d_in[11];
  const float* ff1_b  = (const float*)d_in[12];
  const float* ffln_w = (const float*)d_in[13];
  const float* ffln_b = (const float*)d_in[14];
  const float* ff2_w  = (const float*)d_in[15];
  const float* ff2_b  = (const float*)d_in[16];
  const float* ln2_w  = (const float*)d_in[17];
  const float* ln2_b  = (const float*)d_in[18];
  float* out = (float*)d_out;

  char* ws = (char*)d_ws;
  size_t off = 0;
  auto alloc = [&](size_t bytes) { size_t o = off; off += (bytes + 255) & ~(size_t)255; return o; };

  __hip_bfloat16* CC    = (__hip_bfloat16*)(ws + alloc((size_t)7*65536*2));
  __hip_bfloat16* ff1bf = (__hip_bfloat16*)(ws + alloc((size_t)65536*2));
  __hip_bfloat16* ff2bf = (__hip_bfloat16*)(ws + alloc((size_t)65536*2));
  float* zbias = (float*)(ws + alloc((size_t)256*17*4));
  float* pool  = (float*)(ws + alloc((size_t)2048*NP*4));
  float* THb   = (float*)(ws + alloc((size_t)7*8*128*NP*4));
  float* PHb   = (float*)(ws + alloc((size_t)7*8*128*NP*4));
  float* attnJ = (float*)(ws + alloc((size_t)128*289*4));
  float* attnT = (float*)(ws + alloc((size_t)544*16*4));
  float* attnP = (float*)(ws + alloc((size_t)31232*4));
  float* gstat = (float*)(ws + alloc((size_t)16*4));
  float* ffmu  = (float*)(ws + alloc((size_t)2176*4));
  float* ffrv  = (float*)(ws + alloc((size_t)2176*4));
  float* pS    = (float*)(ws + alloc((size_t)2048*NP*4));
  float* pQ    = (float*)(ws + alloc((size_t)2048*NP*4));
  __hip_bfloat16* Xj  = (__hip_bfloat16*)(ws + alloc(TOTE*2));
  __hip_bfloat16* Xt  = (__hip_bfloat16*)(ws + alloc(TOTE*2));
  __hip_bfloat16* Xv  = (__hip_bfloat16*)(ws + alloc(TOTE*2));
  float* z   = (float*)(ws + alloc(TOTE*4));
  float* zV  = (float*)(ws + alloc(TOTE*4));
  __hip_bfloat16* znb = (__hip_bfloat16*)(ws + alloc(TOTE*2));
  __hip_bfloat16* h1n = Xj;   // Xj is free after the first GEMM
  float* h1 = out;            // use d_out as f32 scratch for the ff1 output
  float* h2 = zV;             // zV is free after k_combine

  const long long sB = (long long)NCIN*NCOL;  // per-batch stride (elements)
  const int h_vbase[5] = {0, 3072, 6144, 9216, 12288};
  const int h_ncol[5]  = {3072, 3072, 3072, 3072, 5120};

  // weight prep
  k_combined_w<<<dim3(256, 7), 256, 0, stream>>>(W_w, g_w, CC);
  k_cvt_bf16<<<256, 256, 0, stream>>>(ff1_w, ff1bf, 65536);
  k_cvt_bf16<<<256, 256, 0, stream>>>(ff2_w, ff2bf, 65536);
  k_zbias<<<1, 256, 0, stream>>>(W_w, g_b, W_b, zbias);

  // attention
  k_pool<<<2048, 256, 0, stream>>>(x, pool);
  k_thph<<<dim3(NP, 56), 128, 0, stream>>>(pool, th_w, th_b, ph_w, ph_b, THb, PHb);
  k_attnJ<<<128, 128, 0, stream>>>(THb, PHb, attnJ);
  k_attnT<<<544, 128, 0, stream>>>(THb, PHb, attnT);
  k_attnP<<<dim3(32, 5), 256, 0, stream>>>(THb, PHb, attnP);
  k_mixJ<<<131072, 128, 0, stream>>>(x, attnJ, Xj);
  k_mixT<<<131072, 128, 0, stream>>>(x, attnT, Xt);
  k_mixV<<<131072, 128, 0, stream>>>(x, attnP, Xv);

  // fused (W · g) projections via bf16 WMMA GEMMs
  k_gemm256<<<dim3(272, 2, NB), 256, 0, stream>>>(CC + 0*65536, Xj, z, nullptr,
                                                  NCOL, NCOL, sB, sB, 0);
  k_gemm256<<<dim3(272, 2, NB), 256, 0, stream>>>(CC + 1*65536, Xt, z, nullptr,
                                                  NCOL, NCOL, sB, sB, 1);
  for (int p = 0; p < 5; ++p) {
    k_gemm256<<<dim3(h_ncol[p]/64, 2, NB), 256, 0, stream>>>(
        CC + (size_t)(2 + p)*65536, Xv + h_vbase[p], zV + h_vbase[p], nullptr,
        NCOL, NCOL, sB, sB, 0);
  }
  k_combine<<<131072, 128, 0, stream>>>(z, zV, zbias, x);

  // GroupNorm 1
  k_zero<<<1, 32, 0, stream>>>(gstat, 16);
  k_gstats<<<2048, 256, 0, stream>>>(z, gstat);
  k_gnorm_dual<<<139264, 256, 0, stream>>>(z, gstat, ln1_w, ln1_b, znb);

  // FFN
  k_gemm256<<<dim3(272, 2, NB), 256, 0, stream>>>(ff1bf, znb, h1, ff1_b,
                                                  NCOL, NCOL, sB, sB, 0);
  k_ffstats1<<<2048, 256, 0, stream>>>(h1, pS, pQ);
  k_ffstats2<<<2176, 256, 0, stream>>>(pS, pQ, ffmu, ffrv);
  k_ffnorm<<<139264, 256, 0, stream>>>(h1, ffmu, ffrv, ffln_w, ffln_b, h1n);
  k_gemm256<<<dim3(272, 2, NB), 256, 0, stream>>>(ff2bf, h1n, h2, ff2_b,
                                                  NCOL, NCOL, sB, sB, 0);

  // residual + GroupNorm 2 -> output
  k_add<<<139264, 256, 0, stream>>>(h2, z);
  k_zero<<<1, 32, 0, stream>>>(gstat, 16);
  k_gstats<<<2048, 256, 0, stream>>>(h2, gstat);
  k_gnorm_out<<<139264, 256, 0, stream>>>(h2, gstat, ln2_w, ln2_b, out);
}
